// get_model_82600811036783
// MI455X (gfx1250) — matrix operand order repaired and checked by reading, not compiled, not run on hardware
//
#include <hip/hip_runtime.h>

typedef __attribute__((ext_vector_type(16))) __bf16          v16bf;
typedef __attribute__((ext_vector_type(16))) unsigned short  v16us;
typedef __attribute__((ext_vector_type(8)))  float           v8f;

#define NB 2
#define N0 16384
#define GEMM_NT 4   // 16x16 N-tiles per wave (padded in packed-W, no guards)

static inline int ceil_div(int a, int b) { return (a + b - 1) / b; }

// ---------------------------------------------------------------------------
// device helpers
// ---------------------------------------------------------------------------
__device__ __forceinline__ unsigned short f32_to_bf16(float f) {
    union { float f; unsigned int u; } c; c.f = f;
    unsigned int u = c.u;
    unsigned int r = u + 0x7fffu + ((u >> 16) & 1u);   // round-to-nearest-even
    return (unsigned short)(r >> 16);
}

// ---------------------------------------------------------------------------
// pack W[K,N] (f32) into bf16 WMMA B-fragment layout, N-tiles padded to a
// multiple of GEMM_NT (phantom tiles are zero):
// P[((tk*ntp + tn)*32 + lane)*16 + e] = bf16(W[tk*32 + hi*8 + (e<8?e:e+8), tn*16 + lr])
// ---------------------------------------------------------------------------
__global__ void pack_w_bf16(const float* __restrict__ W, unsigned short* __restrict__ P,
                            int K, int N)
{
    int kt  = (K + 31) >> 5;
    int nt  = (N + 15) >> 4;
    int ntp = (nt + GEMM_NT - 1) & ~(GEMM_NT - 1);
    int tot = kt * ntp * 512;
    int t = blockIdx.x * blockDim.x + threadIdx.x;
    if (t >= tot) return;
    int e    = t & 15;
    int lane = (t >> 4) & 31;
    int tile = t >> 9;
    int tn = tile % ntp;
    int tk = tile / ntp;
    int hi = lane >> 4, lr = lane & 15;
    int k = (tk << 5) + (hi << 3) + (e < 8 ? e : e + 8);
    int n = (tn << 4) + lr;
    P[t] = (k < K && n < N) ? f32_to_bf16(W[(size_t)k * N + n]) : (unsigned short)0;
}

// ---------------------------------------------------------------------------
// pack A[M,K] (f32) into bf16 WMMA A-fragment layout:
// P[((tm*ktt + tk)*32 + lane)*16 + e] = bf16(A[tm*16 + lr, tk*32 + hi*8 + (e<8?e:e+8)])
// ---------------------------------------------------------------------------
__global__ void pack_a_bf16(const float* __restrict__ A, unsigned short* __restrict__ P,
                            int M, int K)
{
    int mt  = (M + 15) >> 4;
    int ktt = (K + 31) >> 5;
    size_t tot = (size_t)mt * ktt * 512;
    size_t t = (size_t)blockIdx.x * blockDim.x + threadIdx.x;
    if (t >= tot) return;
    int e    = (int)(t & 15);
    int lane = (int)(t >> 4) & 31;
    size_t tile = t >> 9;
    int tk = (int)(tile % (size_t)ktt);
    int tm = (int)(tile / (size_t)ktt);
    int hi = lane >> 4, lr = lane & 15;
    int m = (tm << 4) + lr;
    int k = (tk << 5) + hi * 8 + (e < 8 ? e : e + 8);
    P[t] = (m < M && k < K) ? f32_to_bf16(A[(size_t)m * K + k]) : (unsigned short)0;
}

// ---------------------------------------------------------------------------
// WMMA GEMM: C[M,N] = A[M,K] @ W[K,N] + bias[N]
// Pre-packed bf16 fragments both sides; one wave = 16x64 strip.
// Inner loop is branch-free: 5 fragment loads + 4 v_wmma per K-step.
// ---------------------------------------------------------------------------
__global__ void gemm_bias_wmma(const unsigned short* __restrict__ Ap,
                               const unsigned short* __restrict__ Wp,
                               const float* __restrict__ bias,
                               float* __restrict__ C,
                               int M, int K, int N)
{
    int mt  = (M + 15) >> 4;
    int nt  = (N + 15) >> 4;
    int ntp = (nt + GEMM_NT - 1) & ~(GEMM_NT - 1);
    int ktt = (K + 31) >> 5;
    int ntg = ntp / GEMM_NT;

    // force wave id into an SGPR so tile coords are provably uniform
    int wv   = __builtin_amdgcn_readfirstlane((int)(threadIdx.x >> 5));
    int wave = blockIdx.x * (int)(blockDim.x >> 5) + wv;
    if (wave >= mt * ntg) return;
    int tm  = wave / ntg;
    int tn0 = (wave % ntg) * GEMM_NT;

    int lane = threadIdx.x & 31;
    int lr = lane & 15, hi = lane >> 4;

    const unsigned short* apb = Ap + (size_t)tm * ktt * 512 + (size_t)lane * 16;
    const unsigned short* wpb = Wp + (size_t)tn0 * 512 + (size_t)lane * 16;
    size_t wstride = (size_t)ntp * 512;

    v8f c0 = {}, c1 = {}, c2 = {}, c3 = {};
    #pragma clang loop unroll(disable)
    for (int kt = 0; kt < ktt; ++kt) {
        v16bf av = __builtin_bit_cast(v16bf, *(const v16us*)(apb + (size_t)kt * 512));
        const unsigned short* wp = wpb + (size_t)kt * wstride;
        v16bf b0 = __builtin_bit_cast(v16bf, *(const v16us*)(wp));
        v16bf b1 = __builtin_bit_cast(v16bf, *(const v16us*)(wp + 512));
        v16bf b2 = __builtin_bit_cast(v16bf, *(const v16us*)(wp + 1024));
        v16bf b3 = __builtin_bit_cast(v16bf, *(const v16us*)(wp + 1536));
        c0 = __builtin_amdgcn_wmma_f32_16x16x32_bf16(false, av, false, b0, (short)0, c0, false, false);
        c1 = __builtin_amdgcn_wmma_f32_16x16x32_bf16(false, av, false, b1, (short)0, c1, false, false);
        c2 = __builtin_amdgcn_wmma_f32_16x16x32_bf16(false, av, false, b2, (short)0, c2, false, false);
        c3 = __builtin_amdgcn_wmma_f32_16x16x32_bf16(false, av, false, b3, (short)0, c3, false, false);
    }

    int m0 = tm << 4;
    #pragma unroll
    for (int j = 0; j < GEMM_NT; ++j) {
        v8f acc = (j == 0) ? c0 : (j == 1) ? c1 : (j == 2) ? c2 : c3;
        int col = (tn0 + j) * 16 + lr;
        if (col < N) {
            float bb = bias ? bias[col] : 0.0f;
            #pragma unroll
            for (int r = 0; r < 8; ++r) {
                int rr = m0 + hi * 8 + r;   // lanes 0-15: M=r, 16-31: M=8+r
                if (rr < M) C[(size_t)rr * N + col] = acc[r] + bb;
            }
        }
    }
}

// ---------------------------------------------------------------------------
// per-channel mean / rsqrt(var+eps) over M rows
// ---------------------------------------------------------------------------
__global__ void channel_stats(const float* __restrict__ X, int M, int N,
                              float* __restrict__ mean, float* __restrict__ inv,
                              float eps)
{
    int c = blockIdx.x;
    float s = 0.f, s2 = 0.f;
    for (int r = threadIdx.x; r < M; r += blockDim.x) {
        float v = X[(size_t)r * N + c];
        s += v; s2 += v * v;
    }
    __shared__ float sh[256], sh2[256];
    sh[threadIdx.x] = s; sh2[threadIdx.x] = s2;
    __syncthreads();
    for (int st = blockDim.x >> 1; st > 0; st >>= 1) {
        if ((int)threadIdx.x < st) {
            sh[threadIdx.x]  += sh[threadIdx.x + st];
            sh2[threadIdx.x] += sh2[threadIdx.x + st];
        }
        __syncthreads();
    }
    if (threadIdx.x == 0) {
        float m = sh[0] / (float)M;
        float v = sh2[0] / (float)M - m * m;
        if (v < 0.f) v = 0.f;
        mean[c] = m;
        inv[c]  = rsqrtf(v + eps);
    }
}

__global__ void bn_relu_kernel(float* __restrict__ X, int M, int N,
                               const float* __restrict__ g, const float* __restrict__ be,
                               const float* __restrict__ mean, const float* __restrict__ inv)
{
    size_t tot = (size_t)M * N;
    for (size_t i = (size_t)blockIdx.x * blockDim.x + threadIdx.x; i < tot;
         i += (size_t)gridDim.x * blockDim.x) {
        int c = (int)(i % (size_t)N);
        float v = X[i];
        v = g[c] * (v - mean[c]) * inv[c] + be[c];
        X[i] = v > 0.f ? v : 0.f;
    }
}

// ---------------------------------------------------------------------------
// input transpose: (B,6,N) -> (B,N,6) features + (B,N,3) xyz
// ---------------------------------------------------------------------------
__global__ void prep_input(const float* __restrict__ xin, float* __restrict__ xBNC,
                           float* __restrict__ xyz0, int Nn)
{
    int t = blockIdx.x * blockDim.x + threadIdx.x;
    int tot = NB * Nn * 6;
    if (t >= tot) return;
    int c = t % 6;
    int n = (t / 6) % Nn;
    int b = t / (6 * Nn);
    float v = xin[((size_t)b * 6 + c) * Nn + n];
    xBNC[t] = v;
    if (c < 3) xyz0[((size_t)b * Nn + n) * 3 + c] = v;
}

// ---------------------------------------------------------------------------
// farthest point sampling: one block per batch, sequential npoint iterations
// ---------------------------------------------------------------------------
__global__ void fps_kernel(const float* __restrict__ xyz, int Npts, int npoint,
                           int* __restrict__ fidx, float* __restrict__ distws)
{
    int b = blockIdx.x;
    const float* p = xyz + (size_t)b * Npts * 3;
    float* dist = distws + (size_t)b * Npts;
    __shared__ float sbest[256];
    __shared__ int   sidx[256];
    __shared__ int   sfar;
    for (int i = threadIdx.x; i < Npts; i += blockDim.x) dist[i] = 1e10f;
    if (threadIdx.x == 0) sfar = 0;
    __syncthreads();
    for (int it = 0; it < npoint; ++it) {
        int far = sfar;
        if (threadIdx.x == 0) fidx[b * npoint + it] = far;
        float fx = p[far * 3 + 0], fy = p[far * 3 + 1], fz = p[far * 3 + 2];
        float bd = -1.f; int bi = 0x7fffffff;
        for (int i = threadIdx.x; i < Npts; i += blockDim.x) {
            float dx = p[i * 3 + 0] - fx;
            float dy = p[i * 3 + 1] - fy;
            float dz = p[i * 3 + 2] - fz;
            float d = dx * dx + dy * dy + dz * dz;
            float dm = dist[i];
            if (d < dm) { dm = d; dist[i] = d; }
            if (dm > bd || (dm == bd && i < bi)) { bd = dm; bi = i; }
        }
        sbest[threadIdx.x] = bd; sidx[threadIdx.x] = bi;
        __syncthreads();
        for (int st = blockDim.x >> 1; st > 0; st >>= 1) {
            if ((int)threadIdx.x < st) {
                float od = sbest[threadIdx.x + st]; int oi = sidx[threadIdx.x + st];
                if (od > sbest[threadIdx.x] ||
                    (od == sbest[threadIdx.x] && oi < sidx[threadIdx.x])) {
                    sbest[threadIdx.x] = od; sidx[threadIdx.x] = oi;
                }
            }
            __syncthreads();
        }
        if (threadIdx.x == 0) sfar = sidx[0];
        __syncthreads();
    }
}

// dst[b,s,c] = src[b, idx[b,s], c]
__global__ void gather_points(const float* __restrict__ src, const int* __restrict__ idx,
                              float* __restrict__ dst, int S, int C, int Nsrc)
{
    int t = blockIdx.x * blockDim.x + threadIdx.x;
    int tot = NB * S * C;
    if (t >= tot) return;
    int c = t % C;
    int s = (t / C) % S;
    int b = t / (C * S);
    int id = idx[b * S + s];
    dst[t] = src[((size_t)b * Nsrc + id) * C + c];
}

// first-nsample-in-radius ball query (indices ascending, pad with first)
__global__ void ball_query(const float* __restrict__ xyz, const float* __restrict__ nxyz,
                           int Npts, int S, int ns, float r2, int* __restrict__ gi)
{
    int t = blockIdx.x * blockDim.x + threadIdx.x;
    if (t >= NB * S) return;
    int b = t / S;
    const float* p = xyz + (size_t)b * Npts * 3;
    float cx = nxyz[(size_t)t * 3 + 0];
    float cy = nxyz[(size_t)t * 3 + 1];
    float cz = nxyz[(size_t)t * 3 + 2];
    int* out = gi + (size_t)t * ns;
    int cnt = 0;
    for (int i = 0; i < Npts && cnt < ns; ++i) {
        float dx = p[i * 3 + 0] - cx;
        float dy = p[i * 3 + 1] - cy;
        float dz = p[i * 3 + 2] - cz;
        float d = dx * dx + dy * dy + dz * dz;
        if (d <= r2) out[cnt++] = i;
    }
    int first = (cnt > 0) ? out[0] : 0;
    for (; cnt < ns; ++cnt) out[cnt] = first;
}

// gp[b,s,j,:] = concat(points[b,gi], xyz[b,gi]-new_xyz[b,s])  -> (C+3) channels
__global__ void group_kernel(const float* __restrict__ points, const float* __restrict__ xyz,
                             const float* __restrict__ nxyz, const int* __restrict__ gi,
                             float* __restrict__ gp, int Npts, int S, int ns, int C)
{
    int t = blockIdx.x * blockDim.x + threadIdx.x;
    int tot = NB * S * ns;
    if (t >= tot) return;
    int j = t % ns;
    int s = (t / ns) % S;
    int b = t / (ns * S);
    int id = gi[((size_t)(b * S + s)) * ns + j];
    const float* pf = points + ((size_t)b * Npts + id) * C;
    float* o = gp + (size_t)t * (C + 3);
    for (int c = 0; c < C; ++c) o[c] = pf[c];
    const float* px = xyz + ((size_t)b * Npts + id) * 3;
    const float* nx = nxyz + (size_t)(b * S + s) * 3;
    o[C + 0] = px[0] - nx[0];
    o[C + 1] = px[1] - nx[1];
    o[C + 2] = px[2] - nx[2];
}

// max over ns samples: X (R, ns, C) -> Y[r, cOff+c] in rows of width outC
__global__ void maxpool_kernel(const float* __restrict__ X, float* __restrict__ Y,
                               int R, int ns, int C, int outC, int cOff)
{
    int t = blockIdx.x * blockDim.x + threadIdx.x;
    if (t >= R * C) return;
    int r = t / C, c = t % C;
    const float* x = X + ((size_t)r * ns) * C + c;
    float m = x[0];
    for (int j = 1; j < ns; ++j) {
        float v = x[(size_t)j * C];
        if (v > m) m = v;
    }
    Y[(size_t)r * outC + cOff + c] = m;
}

// feature propagation: out[b,n,:] = concat(pts1[b,n], sum_k w_k * pts2[b,idx_k])
__global__ void fp_interp_concat(const float* __restrict__ xyz1, const float* __restrict__ xyz2,
                                 const float* __restrict__ pts1, const float* __restrict__ pts2,
                                 float* __restrict__ out, int N1, int N2, int C1, int C2)
{
    int t = blockIdx.x * blockDim.x + threadIdx.x;
    if (t >= NB * N1) return;
    int b = t / N1;
    float px = xyz1[(size_t)t * 3 + 0];
    float py = xyz1[(size_t)t * 3 + 1];
    float pz = xyz1[(size_t)t * 3 + 2];
    const float* q = xyz2 + (size_t)b * N2 * 3;
    float d0 = 1e30f, d1 = 1e30f, d2 = 1e30f;
    int   i0 = 0,     i1 = 0,     i2 = 0;
    for (int i = 0; i < N2; ++i) {
        float dx = q[i * 3 + 0] - px;
        float dy = q[i * 3 + 1] - py;
        float dz = q[i * 3 + 2] - pz;
        float d = dx * dx + dy * dy + dz * dz;
        if (d < d0)      { d2 = d1; i2 = i1; d1 = d0; i1 = i0; d0 = d; i0 = i; }
        else if (d < d1) { d2 = d1; i2 = i1; d1 = d; i1 = i; }
        else if (d < d2) { d2 = d; i2 = i; }
    }
    float w0 = 1.f / (d0 + 1e-8f);
    float w1 = 1.f / (d1 + 1e-8f);
    float w2 = 1.f / (d2 + 1e-8f);
    float ws = w0 + w1 + w2;
    w0 /= ws; w1 /= ws; w2 /= ws;

    float* o = out + (size_t)t * (C1 + C2);
    const float* p1 = pts1 + (size_t)t * C1;
    for (int c = 0; c < C1; ++c) o[c] = p1[c];
    const float* base = pts2 + (size_t)b * N2 * C2;
    const float* a0 = base + (size_t)i0 * C2;
    const float* a1 = base + (size_t)i1 * C2;
    const float* a2 = base + (size_t)i2 * C2;
    for (int c = 0; c < C2; ++c)
        o[C1 + c] = w0 * a0[c] + w1 * a1[c] + w2 * a2[c];
}

// (B,N,C) -> (B,C,N)
__global__ void transpose_to_out(const float* __restrict__ X, float* __restrict__ O,
                                 int Nn, int C)
{
    size_t t = (size_t)blockIdx.x * blockDim.x + threadIdx.x;
    size_t tot = (size_t)NB * Nn * C;
    if (t >= tot) return;
    int c = (int)(t % (size_t)C);
    size_t bn = t / (size_t)C;
    int n = (int)(bn % (size_t)Nn);
    int b = (int)(bn / (size_t)Nn);
    O[((size_t)b * C + c) * Nn + n] = X[t];
}

// ---------------------------------------------------------------------------
// host orchestration
// ---------------------------------------------------------------------------
extern "C" void kernel_launch(void* const* d_in, const int* in_sizes, int n_in,
                              void* d_out, int out_size, void* d_ws, size_t ws_size,
                              hipStream_t stream)
{
    (void)in_sizes; (void)n_in; (void)out_size; (void)ws_size;

    auto F = [&](int i) -> const float* { return (const float*)d_in[i]; };

    // ---- workspace bump allocator -----------------------------------------
    size_t off = 0;
    auto alloc = [&](size_t bytes) -> char* {
        char* p = (char*)d_ws + off;
        off += (bytes + 255) & ~(size_t)255;
        return p;
    };
    float* xBNC    = (float*)alloc((size_t)NB * N0 * 6 * 4);
    float* xyz0    = (float*)alloc((size_t)NB * N0 * 3 * 4);
    float* l1_xyz  = (float*)alloc((size_t)NB * 1024 * 3 * 4);
    float* l1_pts  = (float*)alloc((size_t)NB * 1024 * 512 * 4);
    float* l2_xyz  = (float*)alloc((size_t)NB * 512 * 3 * 4);
    float* l2_pts  = (float*)alloc((size_t)NB * 512 * 1024 * 4);
    float* l3_xyz  = (float*)alloc((size_t)NB * 256 * 3 * 4);
    float* l3_pts  = (float*)alloc((size_t)NB * 256 * 2048 * 4);
    float* l0_out  = (float*)alloc((size_t)NB * N0 * 128 * 4);
    int*   fidx    = (int*)  alloc((size_t)NB * 1024 * 4);
    int*   gi      = (int*)  alloc((size_t)NB * 1024 * 64 * 4);
    float* distws  = (float*)alloc((size_t)NB * N0 * 4);
    float* statbuf = (float*)alloc(4096 * 4);
    unsigned short* wpack = (unsigned short*)alloc(8  * 1024 * 1024); // packed W frags
    unsigned short* apack = (unsigned short*)alloc(70 * 1024 * 1024); // packed A frags
    float* bufA    = (float*)alloc(135000064ULL);   // max: sa3 gp (B*256*64*1027*4)
    float* bufB    = (float*)alloc(103000064ULL);   // max: sa3 hidden (B*256*64*784*4)

    // ---- pipeline helpers --------------------------------------------------
    auto launch_gemm = [&](const float* A, const float* W, const float* bias,
                           float* C, int M, int K, int N) {
        int ktt = ceil_div(K, 32), nt = ceil_div(N, 16), mt = ceil_div(M, 16);
        int ntp = (nt + GEMM_NT - 1) & ~(GEMM_NT - 1);
        int wElems = ktt * ntp * 512;
        pack_w_bf16<<<ceil_div(wElems, 256), 256, 0, stream>>>(W, wpack, K, N);
        size_t aElems = (size_t)mt * ktt * 512;
        pack_a_bf16<<<(int)((aElems + 255) / 256), 256, 0, stream>>>(A, apack, M, K);
        int ntg = ntp / GEMM_NT;
        int tiles = mt * ntg;
        gemm_bias_wmma<<<ceil_div(tiles, 8), 256, 0, stream>>>(apack, wpack, bias, C, M, K, N);
    };
    auto mlp_layer = [&](const float* X, float* Y, int pb, int M, int K, int N) {
        // params at pb: W, b, g, be
        launch_gemm(X, F(pb + 0), F(pb + 1), Y, M, K, N);
        channel_stats<<<N, 256, 0, stream>>>(Y, M, N, statbuf, statbuf + N, 1e-5f);
        size_t tot = (size_t)M * N;
        size_t blk = (tot + 255) / 256;
        int blocks = (int)(blk < 4096 ? blk : 4096);
        bn_relu_kernel<<<blocks, 256, 0, stream>>>(Y, M, N, F(pb + 2), F(pb + 3),
                                                   statbuf, statbuf + N);
    };
    auto run_sa = [&](const float* xyz, const float* points, int Nin, int Cin,
                      int npoint, float r0, float r1, int ns0, int ns1,
                      int pbase, int h1, int h2,
                      float* new_xyz, float* new_points) {
        fps_kernel<<<NB, 256, 0, stream>>>(xyz, Nin, npoint, fidx, distws);
        gather_points<<<ceil_div(NB * npoint * 3, 256), 256, 0, stream>>>(
            xyz, fidx, new_xyz, npoint, 3, Nin);
        float rr[2] = {r0, r1};
        int   nv[2] = {ns0, ns1};
        int outC = 2 * h2;
        int cOff = 0;
        for (int br = 0; br < 2; ++br) {
            int ns = nv[br];
            float r2 = rr[br] * rr[br];
            ball_query<<<ceil_div(NB * npoint, 128), 128, 0, stream>>>(
                xyz, new_xyz, Nin, npoint, ns, r2, gi);
            int R = NB * npoint * ns;
            group_kernel<<<ceil_div(R, 256), 256, 0, stream>>>(
                points, xyz, new_xyz, gi, bufA, Nin, npoint, ns, Cin);
            int pb = pbase + br * 8;
            mlp_layer(bufA, bufB, pb + 0, R, Cin + 3, h1);
            mlp_layer(bufB, bufA, pb + 4, R, h1, h2);
            maxpool_kernel<<<ceil_div(NB * npoint * h2, 256), 256, 0, stream>>>(
                bufA, new_points, NB * npoint, ns, h2, outC, cOff);
            cOff += h2;
        }
    };
    auto run_fp = [&](const float* xyz1, const float* xyz2,
                      const float* pts1, const float* pts2,
                      int N1, int N2, int C1, int C2,
                      int pbase, int h1, int h2, float* out) {
        fp_interp_concat<<<ceil_div(NB * N1, 128), 128, 0, stream>>>(
            xyz1, xyz2, pts1, pts2, bufA, N1, N2, C1, C2);
        int M = NB * N1;
        mlp_layer(bufA, bufB, pbase + 0, M, C1 + C2, h1);
        mlp_layer(bufB, out, pbase + 4, M, h1, h2);
    };

    // ---- forward -----------------------------------------------------------
    prep_input<<<ceil_div(NB * N0 * 6, 256), 256, 0, stream>>>(F(0), xBNC, xyz0, N0);

    // param index map (insertion order): sa1=1.. sa2=17.. sa3=33.. fp3=49 fp2=57
    // fp1=65 off1=73 off2=77 dist1=79 dist2=83 cls1=85 cls2=89
    run_sa(xyz0,   xBNC,   N0,   6,    1024, 0.025f, 0.05f, 32, 64,  1, 128, 128,
           l1_xyz, l1_pts);
    run_sa(l1_xyz, l1_pts, 1024, 256,  512,  0.05f,  0.1f,  32, 64, 17, 256, 512,
           l2_xyz, l2_pts);
    run_sa(l2_xyz, l2_pts, 512,  1024, 256,  0.1f,   0.2f,  32, 64, 33, 784, 1024,
           l3_xyz, l3_pts);

    run_fp(l2_xyz, l3_xyz, l2_pts, l3_pts, 512,  256,  1024, 2048, 49, 1024, 1024, l2_pts);
    run_fp(l1_xyz, l2_xyz, l1_pts, l2_pts, 1024, 512,  256,  1024, 57, 512,  512,  l1_pts);
    run_fp(xyz0,   l1_xyz, xBNC,   l1_pts, N0,   1024, 6,    512,  65, 256,  128,  l0_out);

    // ---- outputs -----------------------------------------------------------
    float* out = (float*)d_out;
    size_t o0 = 0;
    size_t o1 = o0 + (size_t)NB * 128 * N0;    // T(l0_points)
    size_t o2 = o1 + (size_t)NB * 2048 * 256;  // T(l3_points)
    size_t o3 = o2 + (size_t)NB * 3 * N0;      // T(l0_xyz)
    size_t o4 = o3 + (size_t)NB * 3 * 256;     // T(l3_xyz)
    size_t o5 = o4 + (size_t)NB * 3 * N0;      // T(off)
    size_t o6 = o5 + (size_t)NB * 1 * N0;      // T(dst)

    transpose_to_out<<<ceil_div(NB * N0 * 128, 256), 256, 0, stream>>>(l0_out, out + o0, N0, 128);
    transpose_to_out<<<ceil_div(NB * 256 * 2048, 256), 256, 0, stream>>>(l3_pts, out + o1, 256, 2048);
    transpose_to_out<<<ceil_div(NB * N0 * 3, 256), 256, 0, stream>>>(xyz0, out + o2, N0, 3);
    transpose_to_out<<<ceil_div(NB * 256 * 3, 256), 256, 0, stream>>>(l3_xyz, out + o3, 256, 3);

    auto run_head = [&](int p1, int wi, int bi, int H, int Cout, float* dst) {
        int M = NB * N0;
        mlp_layer(l0_out, bufA, p1, M, 128, H);            // BN+ReLU hidden
        launch_gemm(bufA, F(wi), F(bi), bufB, M, H, Cout); // plain linear
        transpose_to_out<<<ceil_div(M * Cout, 256), 256, 0, stream>>>(bufB, dst, N0, Cout);
    };
    run_head(73, 77, 78, 16, 3,  out + o4);  // off
    run_head(79, 83, 84, 16, 1,  out + o5);  // dist
    run_head(85, 89, 90, 17, 17, out + o6);  // cls
}
